// Encoder_12618613916304
// MI455X (gfx1250) — matrix operand order repaired and checked by reading, not compiled, not run on hardware
//
#include <hip/hip_runtime.h>

// ---------------------------------------------------------------------------
// Types for CDNA5 WMMA
// ---------------------------------------------------------------------------
typedef __attribute__((ext_vector_type(16))) __bf16 v16bf;
typedef __attribute__((ext_vector_type(8)))  float  v8f;

union ABfrag {
    v16bf v;
    unsigned short s[16];
    uint4 q[2];
};

__device__ __forceinline__ float bf2f(unsigned short b) {
    return __uint_as_float(((unsigned)b) << 16);
}

// ---------------------------------------------------------------------------
// Utility kernels
// ---------------------------------------------------------------------------
__global__ void zero_f32(float* __restrict__ p, long n) {
    long i = (long)blockIdx.x * blockDim.x + threadIdx.x;
    long st = (long)gridDim.x * blockDim.x;
    for (; i < n; i += st) p[i] = 0.0f;
}

// conv1 + conv2 scatter + degree counts in one pass over the edge list.
// d = 64, one thread per (edge, 4 features); lane group 0 also bumps degrees.
__global__ void scatter12(const int* __restrict__ eu, const int* __restrict__ em,
                          const float* __restrict__ xu, const float* __restrict__ xm,
                          float* __restrict__ aggU, float* __restrict__ aggM,
                          float* __restrict__ du, float* __restrict__ dm, int E) {
    int t = blockIdx.x * blockDim.x + threadIdx.x;
    if (t >= E * 16) return;
    int e = t >> 4;
    int c = (t & 15) << 2;
    int u  = eu[e];
    int mv = em[e];
    if ((t & 15) == 0) {                       // one thread per edge: degrees
        atomicAdd(du + u, 1.0f);
        atomicAdd(dm + mv, 1.0f);
    }
    // movie features -> user aggregate
    float4 fm = *(const float4*)(xm + (size_t)mv * 64 + c);
    float* pu = aggU + (size_t)u * 64 + c;
    atomicAdd(pu + 0, fm.x); atomicAdd(pu + 1, fm.y);
    atomicAdd(pu + 2, fm.z); atomicAdd(pu + 3, fm.w);
    // user features -> movie aggregate
    float4 fu = *(const float4*)(xu + (size_t)u * 64 + c);
    float* pm = aggM + (size_t)mv * 64 + c;
    atomicAdd(pm + 0, fu.x); atomicAdd(pm + 1, fu.y);
    atomicAdd(pm + 2, fu.z); atomicAdd(pm + 3, fu.w);
}

// conv3 scatter: movie hidden (bf16, d=128) -> user aggregate (f32)
__global__ void scatter3(const int* __restrict__ eu, const int* __restrict__ em,
                         const unsigned short* __restrict__ h2,
                         float* __restrict__ aggU, int E) {
    int t = blockIdx.x * blockDim.x + threadIdx.x;
    if (t >= E * 32) return;
    int e = t >> 5;
    int c = (t & 31) << 2;
    int u  = eu[e];
    int mv = em[e];
    ushort4 hb = *(const ushort4*)(h2 + (size_t)mv * 128 + c);
    float* p = aggU + (size_t)u * 128 + c;
    atomicAdd(p + 0, bf2f(hb.x)); atomicAdd(p + 1, bf2f(hb.y));
    atomicAdd(p + 2, bf2f(hb.z)); atomicAdd(p + 3, bf2f(hb.w));
}

// ---------------------------------------------------------------------------
// Fused SAGE GEMM:
//   out = act( (agg/deg) @ Wl  +  X @ Wr  + b )        (HASAGG = true)
//   out = X @ Wl + b                                   (HASAGG = false)
// One GEMM with concatenated K' = 2K (or K).
// Block: 256 threads = 8 wave32; wave w -> 16-row M-tile, all NT N-tiles.
// Weights staged transposed in LDS as bf16, padded stride to kill conflicts.
// ---------------------------------------------------------------------------
template<int NT, bool XBF16, bool HASAGG, bool RELU, bool OUTF32>
__global__ __launch_bounds__(256) void sage_gemm(
    int M, int K,
    const float* __restrict__ agg, const float* __restrict__ deg,
    const void* __restrict__ Xv,
    const float* __restrict__ Wl, const float* __restrict__ Wr,
    const float* __restrict__ bias, void* __restrict__ outv)
{
    constexpr int N   = NT * 16;
    constexpr int LDK = 264;                      // 256 max K' + 8 pad (528B pitch)
    __shared__ __align__(32) __bf16 wt[N * LDK];  // W^T as bf16: [n][k]

    const int KT  = HASAGG ? 2 * K : K;
    const int tid = threadIdx.x;

    // Cooperative load of [Wl; Wr] -> transposed bf16 in LDS (hardware cvt)
    for (int i = tid; i < KT * N; i += 256) {
        int k = i / N;
        int n = i - k * N;
        float v = (HASAGG && k >= K) ? Wr[(size_t)(k - K) * N + n]
                                     : Wl[(size_t)k * N + n];
        wt[n * LDK + k] = (__bf16)v;
    }
    __syncthreads();

    const int wave = tid >> 5;
    const int lane = tid & 31;
    const int row0 = blockIdx.x * 128 + wave * 16;
    if (row0 >= M) return;                        // wave-uniform: EXEC stays full

    const int half = lane >> 4;                   // A lane-group select
    const int nl   = lane & 15;
    const int m    = row0 + nl;                   // A row for this lane
    const bool mok = (m < M);

    float sA = 1.0f;
    if (HASAGG) sA = mok ? (1.0f / fmaxf(deg[m], 1.0f)) : 0.0f;

    v8f c[NT];
#pragma unroll
    for (int nt = 0; nt < NT; ++nt)
#pragma unroll
        for (int i = 0; i < 8; ++i) c[nt][i] = 0.0f;

    for (int kc = 0; kc < KT; kc += 32) {
        ABfrag a;
        const bool fromAgg = HASAGG && (kc < K);
        // 16-bit A 16x32 layout: lane half h, elem e -> K = (e>=8?16:0) + 8h + (e&7)
        if (XBF16 && !fromAgg) {
            int kl = HASAGG ? kc - K : kc;
            const unsigned short* p = (const unsigned short*)Xv + (size_t)m * K + kl;
            if (mok) {
                a.q[0] = *(const uint4*)(p + half * 8);
                a.q[1] = *(const uint4*)(p + 16 + half * 8);
            } else {
                a.q[0] = make_uint4(0, 0, 0, 0);
                a.q[1] = make_uint4(0, 0, 0, 0);
            }
        } else {
            const float* p;
            float sc;
            if (fromAgg) { p = agg + (size_t)m * K + kc; sc = sA; }
            else {
                int kl = HASAGG ? kc - K : kc;
                p = (const float*)Xv + (size_t)m * K + kl;
                sc = 1.0f;
            }
            float4 f0, f1, f2, f3;
            if (mok) {
                f0 = *(const float4*)(p + half * 8);
                f1 = *(const float4*)(p + half * 8 + 4);
                f2 = *(const float4*)(p + 16 + half * 8);
                f3 = *(const float4*)(p + 16 + half * 8 + 4);
            } else {
                f0 = f1 = f2 = f3 = make_float4(0, 0, 0, 0);
            }
            a.v[0]  = (__bf16)(f0.x * sc); a.v[1]  = (__bf16)(f0.y * sc);
            a.v[2]  = (__bf16)(f0.z * sc); a.v[3]  = (__bf16)(f0.w * sc);
            a.v[4]  = (__bf16)(f1.x * sc); a.v[5]  = (__bf16)(f1.y * sc);
            a.v[6]  = (__bf16)(f1.z * sc); a.v[7]  = (__bf16)(f1.w * sc);
            a.v[8]  = (__bf16)(f2.x * sc); a.v[9]  = (__bf16)(f2.y * sc);
            a.v[10] = (__bf16)(f2.z * sc); a.v[11] = (__bf16)(f2.w * sc);
            a.v[12] = (__bf16)(f3.x * sc); a.v[13] = (__bf16)(f3.y * sc);
            a.v[14] = (__bf16)(f3.z * sc); a.v[15] = (__bf16)(f3.w * sc);
        }

        // Preload ALL B fragments for this k-chunk, then run the WMMA chain:
        // 16 ds_load_b128 in flight hide LDS latency behind WMMA issue.
        ABfrag b[NT];
#pragma unroll
        for (int nt = 0; nt < NT; ++nt) {
            // 16-bit B 32x16 layout: lane half h, elem e -> K = (e>=8?16:0) + 8h + (e&7)
            const __bf16* bp = &wt[(nt * 16 + nl) * LDK + kc + half * 8];
            b[nt].q[0] = *(const uint4*)bp;
            b[nt].q[1] = *(const uint4*)(bp + 16);
        }
#pragma unroll
        for (int nt = 0; nt < NT; ++nt) {
            c[nt] = __builtin_amdgcn_wmma_f32_16x16x32_bf16(
                        false, a.v, false, b[nt].v, (short)0, c[nt], false, false);
        }
    }

    // C/D layout: VGPR i -> rows row0+i (lanes 0-15) / row0+i+8 (lanes 16-31), col = nl
#pragma unroll
    for (int nt = 0; nt < NT; ++nt) {
        float bv = bias[nt * 16 + nl];
#pragma unroll
        for (int i = 0; i < 8; ++i) {
            int r = row0 + i + half * 8;
            if (r < M) {
                float v = c[nt][i] + bv;
                if (RELU) v = fmaxf(v, 0.0f);
                size_t idx = (size_t)r * N + nt * 16 + nl;
                if (OUTF32) ((float*)outv)[idx] = v;
                else        ((__bf16*)outv)[idx] = (__bf16)v;
            }
        }
    }
}

// ---------------------------------------------------------------------------
// Host-side launch
// ---------------------------------------------------------------------------
extern "C" void kernel_launch(void* const* d_in, const int* in_sizes, int n_in,
                              void* d_out, int out_size, void* d_ws, size_t ws_size,
                              hipStream_t stream) {
    (void)in_sizes; (void)n_in; (void)out_size; (void)ws_size;

    constexpr int NU = 100000, NM = 20000, DIN = 64, HID = 128, OUT = 64, E = 600000;

    const float* x_user     = (const float*)d_in[0];
    const float* x_movie    = (const float*)d_in[1];
    const int*   edge_user  = (const int*)  d_in[2];
    const int*   edge_movie = (const int*)  d_in[3];
    const float* W1l = (const float*)d_in[4];
    const float* W1r = (const float*)d_in[5];
    const float* b1  = (const float*)d_in[6];
    const float* W2l = (const float*)d_in[7];
    const float* W2r = (const float*)d_in[8];
    const float* b2  = (const float*)d_in[9];
    const float* W3l = (const float*)d_in[10];
    const float* W3r = (const float*)d_in[11];
    const float* b3  = (const float*)d_in[12];
    const float* Wlin1 = (const float*)d_in[13];
    const float* blin1 = (const float*)d_in[14];
    const float* Wlin2 = (const float*)d_in[15];
    const float* blin2 = (const float*)d_in[16];

    float* out_user  = (float*)d_out;
    float* out_movie = out_user + (size_t)NU * OUT;

    // workspace carve-up (256B aligned slabs)
    char* ws = (char*)d_ws;
    size_t off = 0;
    auto alloc = [&](size_t bytes) -> void* {
        void* p = ws + off;
        off = (off + bytes + 255) & ~(size_t)255;
        return p;
    };
    float*          deg_u = (float*)alloc((size_t)NU * 4);
    float*          deg_m = (float*)alloc((size_t)NM * 4);
    float*          aggU  = (float*)alloc((size_t)NU * HID * 4);  // [NUx64] then [NUx128]
    float*          aggM  = (float*)alloc((size_t)NM * DIN * 4);
    unsigned short* h1u   = (unsigned short*)alloc((size_t)NU * HID * 2);
    unsigned short* h2m   = (unsigned short*)alloc((size_t)NM * HID * 2);
    unsigned short* h3u   = (unsigned short*)alloc((size_t)NU * HID * 2);

    // ---- zero accumulators / degrees ----
    zero_f32<<<2048, 256, 0, stream>>>(deg_u, (long)NU);
    zero_f32<<<2048, 256, 0, stream>>>(deg_m, (long)NM);
    zero_f32<<<4096, 256, 0, stream>>>(aggU, (long)NU * DIN);
    zero_f32<<<4096, 256, 0, stream>>>(aggM, (long)NM * DIN);

    // ---- conv1/conv2 scatters (+ degrees fused) ----
    scatter12<<<(E * 16 + 255) / 256, 256, 0, stream>>>(edge_user, edge_movie,
                                                        x_user, x_movie, aggU, aggM,
                                                        deg_u, deg_m, E);

    // ---- conv1: user hidden ----
    sage_gemm<8, false, true, true, false><<<(NU + 127) / 128, 256, 0, stream>>>(
        NU, DIN, aggU, deg_u, x_user, W1l, W1r, b1, h1u);
    // ---- conv2: movie hidden ----
    sage_gemm<8, false, true, true, false><<<(NM + 127) / 128, 256, 0, stream>>>(
        NM, DIN, aggM, deg_m, x_movie, W2l, W2r, b2, h2m);

    // ---- conv3 aggregate (reuse aggU as [NU x 128]) ----
    zero_f32<<<4096, 256, 0, stream>>>(aggU, (long)NU * HID);
    scatter3<<<(E * 32 + 255) / 256, 256, 0, stream>>>(edge_user, edge_movie, h2m, aggU, E);

    // ---- conv3: user hidden (X is bf16) ----
    sage_gemm<8, true, true, true, false><<<(NU + 127) / 128, 256, 0, stream>>>(
        NU, HID, aggU, deg_u, h1u, W3l, W3r, b3, h3u);

    // ---- linear heads (plain GEMM, f32 out) ----
    sage_gemm<4, true, false, false, true><<<(NU + 127) / 128, 256, 0, stream>>>(
        NU, HID, nullptr, nullptr, h3u, Wlin1, nullptr, blin1, out_user);
    sage_gemm<4, true, false, false, true><<<(NM + 127) / 128, 256, 0, stream>>>(
        NM, HID, nullptr, nullptr, h2m, Wlin2, nullptr, blin2, out_movie);
}
